// Value_65764539236871
// MI455X (gfx1250) — matrix operand order repaired and checked by reading, not compiled, not run on hardware
//
#include <hip/hip_runtime.h>
#include <math.h>

typedef __attribute__((ext_vector_type(16))) __bf16 v16bf;
typedef __attribute__((ext_vector_type(8)))  __bf16 v8bf;
typedef __attribute__((ext_vector_type(8)))  float  v8f;

#define D_MODEL 512
#define NROWS_ACT 16384   // B*S*S = 4*64*64
#define NROWS_OBS 256     // B*S

__device__ __forceinline__ float gelu_tanh(float x) {
    float x3 = x * x * x;
    return 0.5f * x * (1.0f + tanhf(0.7978845608028654f * (x + 0.044715f * x3)));
}

// ---------------------------------------------------------------------------
// Pack fp32 weight [K,N] row-major into bf16 WMMA B-fragment order:
// idx = ((nt*kTiles + kt)*32 + lane)*16 + e  where k = kt*32 + (lane>>4)*16 + e,
// n = nt*16 + (lane&15).  One contiguous 32B load per lane per (kt,nt) in GEMM.
// ---------------------------------------------------------------------------
__global__ void pack_w_kernel(const float* __restrict__ W, __bf16* __restrict__ Wp,
                              int K, int N) {
    int i = blockIdx.x * 256 + threadIdx.x;
    if (i >= K * N) return;
    int e    = i & 15;
    int lane = (i >> 4) & 31;
    int t    = i >> 9;           // nt*kTiles + kt
    int kTiles = K >> 5;
    int kt = t % kTiles;
    int nt = t / kTiles;
    int k = kt * 32 + (lane >> 4) * 16 + e;
    int n = nt * 16 + (lane & 15);
    Wp[i] = (__bf16)W[(size_t)k * N + n];
}

// ---------------------------------------------------------------------------
// bf16 WMMA GEMM: out[M,N] = A[M,K] @ W[K,N] + bias, with fused epilogues.
// Workgroup = 8 waves; wave w handles rows [blk*128 + 16w, +16), cols [64].
// ---------------------------------------------------------------------------
enum { EPI_F32 = 0, EPI_GELU_BF16 = 1, EPI_RESID_F32 = 2 };

template <int EPI>
__global__ __launch_bounds__(256) void gemm_bf16_wmma(
    const __bf16* __restrict__ A, const __bf16* __restrict__ Wp,
    const float* __restrict__ bias, float* __restrict__ outF,
    __bf16* __restrict__ outH, const float* __restrict__ resid,
    int N, int K) {
    const int wave = threadIdx.x >> 5;
    const int lane = threadIdx.x & 31;
    const int half = lane >> 4;
    const int l15  = lane & 15;
    const int rowBase = blockIdx.x * 128 + wave * 16;
    const int colBase = blockIdx.y * 64;
    const int kTiles = K >> 5;

    v8f acc[4] = {v8f{}, v8f{}, v8f{}, v8f{}};
    const __bf16* __restrict__ arow  = A + (size_t)(rowBase + l15) * K + half * 16;
    const __bf16* __restrict__ bbase = Wp + (size_t)(colBase >> 4) * kTiles * 512
                                          + (size_t)lane * 16;

    for (int kt = 0; kt < kTiles; ++kt) {
        v8bf lo = *(const v8bf*)(arow + kt * 32);        // K = k0..k0+7   (+half*16)
        v8bf hi = *(const v8bf*)(arow + kt * 32 + 8);    // K = k0+8..+15  (+half*16)
        v16bf a = __builtin_shufflevector(lo, hi, 0, 1, 2, 3, 4, 5, 6, 7,
                                          8, 9, 10, 11, 12, 13, 14, 15);
#pragma unroll
        for (int j = 0; j < 4; ++j) {
            v16bf b = *(const v16bf*)(bbase + ((size_t)j * kTiles + kt) * 512);
            acc[j] = __builtin_amdgcn_wmma_f32_16x16x32_bf16(
                false, a, false, b, (short)0, acc[j], false, false);
        }
    }
#pragma unroll
    for (int j = 0; j < 4; ++j) {
        const int col = colBase + j * 16 + l15;
        const float bv = bias[col];
#pragma unroll
        for (int v = 0; v < 8; ++v) {
            const int row = rowBase + half * 8 + v;   // C/D layout: VGPR v -> M=v(+8)
            const size_t idx = (size_t)row * N + col;
            float val = acc[j][v] + bv;
            if (EPI == EPI_GELU_BF16)      outH[idx] = (__bf16)gelu_tanh(val);
            else if (EPI == EPI_RESID_F32) outF[idx] = resid[idx] + val;
            else                           outF[idx] = val;
        }
    }
}

// ---------------------------------------------------------------------------
// LayerNorm (512 cols) -> bf16
// ---------------------------------------------------------------------------
__global__ __launch_bounds__(128) void ln_bf16_kernel(
    const float* __restrict__ x, const float* __restrict__ bias,
    const float* __restrict__ scale, __bf16* __restrict__ out) {
    __shared__ float red[128];
    const int row = blockIdx.x, tid = threadIdx.x;
    const float4 v = ((const float4*)(x + (size_t)row * 512))[tid];
    red[tid] = v.x + v.y + v.z + v.w;
    __syncthreads();
    for (int st = 64; st; st >>= 1) { if (tid < st) red[tid] += red[tid + st]; __syncthreads(); }
    const float mean = red[0] * (1.0f / 512.0f);
    __syncthreads();
    const float dx = v.x - mean, dy = v.y - mean, dz = v.z - mean, dw = v.w - mean;
    red[tid] = dx * dx + dy * dy + dz * dz + dw * dw;
    __syncthreads();
    for (int st = 64; st; st >>= 1) { if (tid < st) red[tid] += red[tid + st]; __syncthreads(); }
    const float inv = rsqrtf(red[0] * (1.0f / 512.0f) + 1e-6f);
    const int c = tid * 4;
    __bf16* o = out + (size_t)row * 512 + c;
    o[0] = (__bf16)(dx * inv * scale[c + 0] + bias[c + 0]);
    o[1] = (__bf16)(dy * inv * scale[c + 1] + bias[c + 1]);
    o[2] = (__bf16)(dz * inv * scale[c + 2] + bias[c + 2]);
    o[3] = (__bf16)(dw * inv * scale[c + 3] + bias[c + 3]);
}

// ---------------------------------------------------------------------------
// Final LN + q_value head (D->1), one block per act row
// ---------------------------------------------------------------------------
__global__ __launch_bounds__(128) void ln_head_kernel(
    const float* __restrict__ x, const float* __restrict__ lbias,
    const float* __restrict__ lscale, const float* __restrict__ Wq,
    const float* __restrict__ bq, float* __restrict__ out) {
    __shared__ float red[128];
    const int row = blockIdx.x, tid = threadIdx.x;
    const float4 v = ((const float4*)(x + (size_t)row * 512))[tid];
    red[tid] = v.x + v.y + v.z + v.w;
    __syncthreads();
    for (int st = 64; st; st >>= 1) { if (tid < st) red[tid] += red[tid + st]; __syncthreads(); }
    const float mean = red[0] * (1.0f / 512.0f);
    __syncthreads();
    const float dx = v.x - mean, dy = v.y - mean, dz = v.z - mean, dw = v.w - mean;
    red[tid] = dx * dx + dy * dy + dz * dz + dw * dw;
    __syncthreads();
    for (int st = 64; st; st >>= 1) { if (tid < st) red[tid] += red[tid + st]; __syncthreads(); }
    const float inv = rsqrtf(red[0] * (1.0f / 512.0f) + 1e-6f);
    __syncthreads();
    const int c = tid * 4;
    red[tid] = (dx * inv * lscale[c + 0] + lbias[c + 0]) * Wq[c + 0]
             + (dy * inv * lscale[c + 1] + lbias[c + 1]) * Wq[c + 1]
             + (dz * inv * lscale[c + 2] + lbias[c + 2]) * Wq[c + 2]
             + (dw * inv * lscale[c + 3] + lbias[c + 3]) * Wq[c + 3];
    __syncthreads();
    for (int st = 64; st; st >>= 1) { if (tid < st) red[tid] += red[tid + st]; __syncthreads(); }
    if (tid == 0) out[row] = red[0] + bq[0];
}

// ---------------------------------------------------------------------------
// Attention: one block per (b, obs, head), 64 threads = queries.
// Keys: index 0 = obs key (always visible); key j(1..64) = act slot j-1,
// visible iff o <= j-1 <= q.  Adds result into act_resid.
// ---------------------------------------------------------------------------
__global__ __launch_bounds__(64) void attn_kernel(
    const float* __restrict__ qb, const float* __restrict__ kact,
    const float* __restrict__ vact, const float* __restrict__ kobs,
    const float* __restrict__ vobs, float* __restrict__ act_resid) {
    __shared__ float sk[65][64];
    __shared__ float sv[65][64];
    __shared__ float sp[64][66];
    const int bid = blockIdx.x;
    const int h = bid & 7, o = (bid >> 3) & 63, b = bid >> 9;
    const int tid = threadIdx.x;
    const int bo = b * 64 + o;

    for (int idx = tid; idx < 65 * 64; idx += 64) {
        const int r = idx >> 6, c = idx & 63;
        if (r == 0) {
            sk[0][c] = kobs[(size_t)bo * 512 + h * 64 + c];
            sv[0][c] = vobs[(size_t)bo * 512 + h * 64 + c];
        } else {
            const size_t ri = ((size_t)bo * 64 + (r - 1)) * 512 + h * 64 + c;
            sk[r][c] = kact[ri];
            sv[r][c] = vact[ri];
        }
    }
    float qr[64];
    const size_t qi = ((size_t)bo * 64 + tid) * 512 + h * 64;
#pragma unroll
    for (int d = 0; d < 64; ++d) qr[d] = qb[qi + d];
    __syncthreads();

    for (int j = 0; j < 65; ++j) {
        float s = 0.0f;
#pragma unroll
        for (int d = 0; d < 64; ++d) s += qr[d] * sk[j][d];
        s *= 0.125f;  // 1/sqrt(64)
        const bool vis = (j == 0) || ((j - 1) >= o && (j - 1) <= tid);
        sp[tid][j] = vis ? s : -1e6f;
    }
    float m = -1e30f;
    for (int j = 0; j < 65; ++j) m = fmaxf(m, sp[tid][j]);
    float sum = 0.0f;
    for (int j = 0; j < 65; ++j) { float p = expf(sp[tid][j] - m); sp[tid][j] = p; sum += p; }
    const float inv = 1.0f / sum;

    float outr[64];
#pragma unroll
    for (int d = 0; d < 64; ++d) outr[d] = 0.0f;
    for (int j = 0; j < 65; ++j) {
        const float p = sp[tid][j] * inv;
#pragma unroll
        for (int d = 0; d < 64; ++d) outr[d] += p * sv[j][d];
    }
#pragma unroll
    for (int d = 0; d < 64; ++d) act_resid[qi + d] += outr[d];
}

// ---------------------------------------------------------------------------
// Embedding kernels
// ---------------------------------------------------------------------------
__global__ void obs_embed_kernel(const float* __restrict__ obs, const float* __restrict__ W,
                                 const float* __restrict__ b, float* __restrict__ out) {
    const int i = blockIdx.x * 256 + threadIdx.x;  // 256*512 threads
    const int n = i & 511, row = i >> 9;
    const float* xr = obs + (size_t)row * 128;
    float s = b[n];
#pragma unroll 4
    for (int c = 0; c < 128; ++c) s += xr[c] * W[(size_t)c * 512 + n];
    out[i] = s;
}

__global__ void act_embed_kernel(const float* __restrict__ act, const float* __restrict__ W,
                                 const float* __restrict__ b, float* __restrict__ out) {
    const int i = blockIdx.x * 256 + threadIdx.x;  // 256*512 threads
    const int n = i & 511, row = i >> 9;
    const float* xr = act + (size_t)row * 32;
    float s = b[n];
#pragma unroll
    for (int c = 0; c < 32; ++c) s += xr[c] * W[(size_t)c * 512 + n];
    out[i] = s;
}

// act_resid[b,s1,s2,c] = act_embed[b,s1,c] + pos_embed[max(s2-s1,0), c]
__global__ void act_broadcast_kernel(const float* __restrict__ emb,
                                     const float* __restrict__ pos,
                                     float* __restrict__ out) {
    const int i = blockIdx.x * 256 + threadIdx.x;  // 4*64*64*512 threads
    const int c  = i & 511;
    const int s2 = (i >> 9) & 63;
    const int s1 = (i >> 15) & 63;
    const int b  = i >> 21;
    int idx = s2 - s1; if (idx < 0) idx = 0;
    out[i] = emb[(((size_t)b * 64 + s1) << 9) + c] + pos[((size_t)idx << 9) + c];
}

// ---------------------------------------------------------------------------
// Host orchestration
// ---------------------------------------------------------------------------
extern "C" void kernel_launch(void* const* d_in, const int* in_sizes, int n_in,
                              void* d_out, int out_size, void* d_ws, size_t ws_size,
                              hipStream_t stream) {
    (void)in_sizes; (void)n_in; (void)out_size; (void)ws_size;
    auto F = [&](int i) { return (const float*)d_in[i]; };

    // Workspace carve (256B aligned)
    size_t off = 0;
    auto carve = [&](size_t bytes) {
        void* p = (char*)d_ws + off;
        off += (bytes + 255) & ~(size_t)255;
        return p;
    };
    float* act_resid = (float*)carve((size_t)NROWS_ACT * 512 * 4);
    float* obs_resid = (float*)carve((size_t)NROWS_OBS * 512 * 4);
    float* act_emb   = (float*)carve((size_t)NROWS_OBS * 512 * 4);
    __bf16* actln    = (__bf16*)carve((size_t)NROWS_ACT * 512 * 2);
    __bf16* obsln    = (__bf16*)carve((size_t)NROWS_OBS * 512 * 2);
    float* qbuf      = (float*)carve((size_t)NROWS_ACT * 512 * 4);
    float* kbuf      = (float*)carve((size_t)NROWS_ACT * 512 * 4);
    float* vbuf      = (float*)carve((size_t)NROWS_ACT * 512 * 4);
    float* kobs      = (float*)carve((size_t)NROWS_OBS * 512 * 4);
    float* vobs      = (float*)carve((size_t)NROWS_OBS * 512 * 4);
    __bf16* hidden_act = (__bf16*)qbuf;  // alias: 64MB over qbuf+kbuf (attn done before MLP)
    __bf16* hidden_obs = (__bf16*)carve((size_t)NROWS_OBS * 2048 * 2);
    __bf16* p_q   = (__bf16*)carve(512 * 512 * 2);
    __bf16* p_ka  = (__bf16*)carve(512 * 512 * 2);
    __bf16* p_va  = (__bf16*)carve(512 * 512 * 2);
    __bf16* p_ko  = (__bf16*)carve(512 * 512 * 2);
    __bf16* p_vo  = (__bf16*)carve(512 * 512 * 2);
    __bf16* p_am1 = (__bf16*)carve((size_t)512 * 2048 * 2);
    __bf16* p_am2 = (__bf16*)carve((size_t)2048 * 512 * 2);
    __bf16* p_om1 = (__bf16*)carve((size_t)512 * 2048 * 2);
    __bf16* p_om2 = (__bf16*)carve((size_t)2048 * 512 * 2);

    // ---- Embeddings ----
    obs_embed_kernel<<<(NROWS_OBS * 512) / 256, 256, 0, stream>>>(F(0), F(86), F(87), obs_resid);
    act_embed_kernel<<<(NROWS_OBS * 512) / 256, 256, 0, stream>>>(F(1), F(3), F(4), act_emb);
    act_broadcast_kernel<<<(NROWS_ACT * 512) / 256, 256, 0, stream>>>(act_emb, F(5), act_resid);

    // ---- Layers ----
    for (int l = 0; l < 3; ++l) {
        const int base = 8 + 26 * l;
        // per-layer leaf offsets (sorted-key pytree order):
        // +0 act_key.W  +1 act_key.b  +2 act_ln1.bias +3 act_ln1.scale
        // +4 act_ln2.bias +5 act_ln2.scale +6 act_mlp_1.W +7 act_mlp_1.b
        // +8 act_mlp_2.W +9 act_mlp_2.b +10 act_query.W +11 act_query.b
        // +12 act_value.W +13 act_value.b +14 obs_key.W +15 obs_key.b
        // +16 obs_ln1.bias +17 obs_ln1.scale +18 obs_ln2.bias +19 obs_ln2.scale
        // +20 obs_mlp_1.W +21 obs_mlp_1.b +22 obs_mlp_2.W +23 obs_mlp_2.b
        // +24 obs_value.W +25 obs_value.b
        pack_w_kernel<<<(512 * 512) / 256, 256, 0, stream>>>(F(base + 10), p_q, 512, 512);
        pack_w_kernel<<<(512 * 512) / 256, 256, 0, stream>>>(F(base + 0), p_ka, 512, 512);
        pack_w_kernel<<<(512 * 512) / 256, 256, 0, stream>>>(F(base + 12), p_va, 512, 512);
        pack_w_kernel<<<(512 * 512) / 256, 256, 0, stream>>>(F(base + 14), p_ko, 512, 512);
        pack_w_kernel<<<(512 * 512) / 256, 256, 0, stream>>>(F(base + 24), p_vo, 512, 512);
        pack_w_kernel<<<(512 * 2048) / 256, 256, 0, stream>>>(F(base + 6), p_am1, 512, 2048);
        pack_w_kernel<<<(2048 * 512) / 256, 256, 0, stream>>>(F(base + 8), p_am2, 2048, 512);
        pack_w_kernel<<<(512 * 2048) / 256, 256, 0, stream>>>(F(base + 20), p_om1, 512, 2048);
        pack_w_kernel<<<(2048 * 512) / 256, 256, 0, stream>>>(F(base + 22), p_om2, 2048, 512);

        // LN1
        ln_bf16_kernel<<<NROWS_ACT, 128, 0, stream>>>(act_resid, F(base + 2), F(base + 3), actln);
        ln_bf16_kernel<<<NROWS_OBS, 128, 0, stream>>>(obs_resid, F(base + 16), F(base + 17), obsln);

        // Q/K/V projections (WMMA)
        gemm_bf16_wmma<EPI_F32><<<dim3(NROWS_ACT / 128, 8), 256, 0, stream>>>(
            actln, p_q, F(base + 11), qbuf, nullptr, nullptr, 512, 512);
        gemm_bf16_wmma<EPI_F32><<<dim3(NROWS_ACT / 128, 8), 256, 0, stream>>>(
            actln, p_ka, F(base + 1), kbuf, nullptr, nullptr, 512, 512);
        gemm_bf16_wmma<EPI_F32><<<dim3(NROWS_ACT / 128, 8), 256, 0, stream>>>(
            actln, p_va, F(base + 13), vbuf, nullptr, nullptr, 512, 512);
        gemm_bf16_wmma<EPI_F32><<<dim3(NROWS_OBS / 128, 8), 256, 0, stream>>>(
            obsln, p_ko, F(base + 15), kobs, nullptr, nullptr, 512, 512);
        gemm_bf16_wmma<EPI_F32><<<dim3(NROWS_OBS / 128, 8), 256, 0, stream>>>(
            obsln, p_vo, F(base + 25), vobs, nullptr, nullptr, 512, 512);

        // Attention (adds into act_resid)
        attn_kernel<<<4 * 64 * 8, 64, 0, stream>>>(qbuf, kbuf, vbuf, kobs, vobs, act_resid);

        // LN2
        ln_bf16_kernel<<<NROWS_ACT, 128, 0, stream>>>(act_resid, F(base + 4), F(base + 5), actln);
        ln_bf16_kernel<<<NROWS_OBS, 128, 0, stream>>>(obs_resid, F(base + 18), F(base + 19), obsln);

        // MLPs (WMMA, gelu fused in GEMM1 epilogue, residual fused in GEMM2)
        gemm_bf16_wmma<EPI_GELU_BF16><<<dim3(NROWS_ACT / 128, 32), 256, 0, stream>>>(
            actln, p_am1, F(base + 7), nullptr, hidden_act, nullptr, 2048, 512);
        gemm_bf16_wmma<EPI_RESID_F32><<<dim3(NROWS_ACT / 128, 8), 256, 0, stream>>>(
            hidden_act, p_am2, F(base + 9), act_resid, nullptr, act_resid, 512, 2048);
        gemm_bf16_wmma<EPI_GELU_BF16><<<dim3(NROWS_OBS / 128, 32), 256, 0, stream>>>(
            obsln, p_om1, F(base + 21), nullptr, hidden_obs, nullptr, 2048, 512);
        gemm_bf16_wmma<EPI_RESID_F32><<<dim3(NROWS_OBS / 128, 8), 256, 0, stream>>>(
            hidden_obs, p_om2, F(base + 23), obs_resid, nullptr, obs_resid, 512, 2048);
    }

    // ---- Final LN + head ----
    ln_head_kernel<<<NROWS_ACT, 128, 0, stream>>>(act_resid, F(6), F(7), F(88), F(89),
                                                  (float*)d_out);
}
